// diff_RATD_22840636080886
// MI455X (gfx1250) — matrix operand order repaired and checked by reading, not compiled, not run on hardware
//
#include <hip/hip_runtime.h>
#include <math.h>

// ---------------- CDNA5 WMMA types & fragment helpers ----------------
typedef __bf16 bf16_t;
typedef __bf16 v16bf __attribute__((ext_vector_type(16)));
typedef __bf16 v4bf  __attribute__((ext_vector_type(4)));
typedef float  v8f   __attribute__((ext_vector_type(8)));

static __device__ __forceinline__ v8f wmma_bf16(v16bf a, v16bf b, v8f c) {
    // D = A(16x32) * B(32x16) + C, fp32 accumulate
    return __builtin_amdgcn_wmma_f32_16x16x32_bf16(false, a, false, b, (short)0, c, false, false);
}

// A fragment: LDS tile stored row-major (m, k), ld halves per row.
// Per ISA 7.12.2: lane<16 holds row (lane), K in {0..7, 16..23}; lane>=16 holds K in {8..15, 24..31}.
static __device__ __forceinline__ v16bf load_a_frag(const bf16_t* lds, int row_base, int ld,
                                                    int k_base, int lane) {
    int r  = row_base + (lane & 15);
    int hi = (lane >> 4) & 1;
    const bf16_t* p0 = lds + r * ld + k_base + hi * 8;
    v16bf f;
#pragma unroll
    for (int i = 0; i < 8; i++) f[i] = p0[i];
#pragma unroll
    for (int i = 0; i < 8; i++) f[i + 8] = p0[16 + i];
    return f;
}

// B fragment: LDS tile stored (n, k) row-major; lane col = lane&15, K in {0..7,16..23} (lane<16) / {8..15,24..31} (lane>=16).
static __device__ __forceinline__ v16bf load_b_frag(const bf16_t* lds, int col_base, int ld,
                                                    int k_base, int lane) {
    int n  = col_base + (lane & 15);
    int hi = (lane >> 4) & 1;
    const bf16_t* p = lds + n * ld + k_base + hi * 8;
    v16bf f;
#pragma unroll
    for (int i = 0; i < 8; i++) { f[i] = p[i]; f[i + 8] = p[16 + i]; }
    return f;
}

// ---------------- Generic fused GEMM:  D = act(ascale*A (MxK) * W(NxK)^T + bias) ----------------
#define BM 128
#define BN 64
#define BK 32

// Fully in-bounds tile staging: float4 global loads + packed v4bf LDS stores.
static __device__ __forceinline__ void stage_fast(
    const float* __restrict__ G, bf16_t* __restrict__ L,
    int rows, int r0, int K, int k0, float scale, int tid)
{
    for (int i = tid * 4; i < rows * BK; i += 256 * 4) {
        int row = i >> 5, col = i & 31;
        const float4 f = *(const float4*)(G + (long long)(r0 + row) * K + k0 + col);
        v4bf h;
        h[0] = (bf16_t)(f.x * scale);
        h[1] = (bf16_t)(f.y * scale);
        h[2] = (bf16_t)(f.z * scale);
        h[3] = (bf16_t)(f.w * scale);
        *(v4bf*)(L + row * BK + col) = h;
    }
}

// Guarded scalar staging for ragged edges (zero-fill OOB).
static __device__ __forceinline__ void stage_guard(
    const float* __restrict__ G, bf16_t* __restrict__ L,
    int rows, int r0, int Rlim, int K, int k0, float scale, int tid)
{
    for (int i = tid * 4; i < rows * BK; i += 256 * 4) {
        int row = i >> 5, col = i & 31;
#pragma unroll
        for (int j = 0; j < 4; j++) {
            int gr = r0 + row, gk = k0 + col + j;
            float v = (gr < Rlim && gk < K) ? G[(long long)gr * K + gk] * scale : 0.f;
            L[row * BK + col + j] = (bf16_t)v;
        }
    }
}

template <bool AL>
__global__ __launch_bounds__(256) void gemm_bf16(
    const float* __restrict__ A, const float* __restrict__ W,
    const float* __restrict__ bias, float* __restrict__ D,
    int M, int N, int K, int act, float ascale)
{
    __shared__ bf16_t Alds[2][BM * BK];
    __shared__ bf16_t Blds[2][BN * BK];

    const int m0 = blockIdx.y * BM;
    const int n0 = blockIdx.x * BN;
    const int tid = threadIdx.x;
    const int lane = tid & 31;
    const int wave = tid >> 5;      // 8 waves
    const int wm = wave >> 1;       // 0..3 -> 32-row slab
    const int wn = wave & 1;        // 0..1 -> 32-col slab

    const int nk = (K + BK - 1) / BK;

    v8f acc[2][2] = {};

    // prologue: stage tile 0
    if constexpr (AL) {
        stage_fast(A, Alds[0], BM, m0, K, 0, ascale, tid);
        stage_fast(W, Blds[0], BN, n0, K, 0, 1.f, tid);
    } else {
        const bool fA = (m0 + BM <= M) && (BK <= K);
        const bool fB = (n0 + BN <= N) && (BK <= K);
        if (fA) stage_fast(A, Alds[0], BM, m0, K, 0, ascale, tid);
        else    stage_guard(A, Alds[0], BM, m0, M, K, 0, ascale, tid);
        if (fB) stage_fast(W, Blds[0], BN, n0, K, 0, 1.f, tid);
        else    stage_guard(W, Blds[0], BN, n0, N, K, 0, 1.f, tid);
    }
    __syncthreads();

    int buf = 0;
    for (int t = 0; t < nk; t++, buf ^= 1) {
        // prefetch next tile into the other LDS buffer (overlaps with WMMA below)
        if (t + 1 < nk) {
            int k0n = (t + 1) * BK;
            if constexpr (AL) {
                stage_fast(A, Alds[buf ^ 1], BM, m0, K, k0n, ascale, tid);
                stage_fast(W, Blds[buf ^ 1], BN, n0, K, k0n, 1.f, tid);
            } else {
                const bool fK = (k0n + BK <= K);
                if ((m0 + BM <= M) && fK) stage_fast(A, Alds[buf ^ 1], BM, m0, K, k0n, ascale, tid);
                else                      stage_guard(A, Alds[buf ^ 1], BM, m0, M, K, k0n, ascale, tid);
                if ((n0 + BN <= N) && fK) stage_fast(W, Blds[buf ^ 1], BN, n0, K, k0n, 1.f, tid);
                else                      stage_guard(W, Blds[buf ^ 1], BN, n0, N, K, k0n, 1.f, tid);
            }
        }
#pragma unroll
        for (int tm = 0; tm < 2; tm++) {
            v16bf af = load_a_frag(Alds[buf], wm * 32 + tm * 16, BK, 0, lane);
#pragma unroll
            for (int tn = 0; tn < 2; tn++) {
                v16bf bf_ = load_b_frag(Blds[buf], wn * 32 + tn * 16, BK, 0, lane);
                acc[tm][tn] = wmma_bf16(af, bf_, acc[tm][tn]);
            }
        }
        __syncthreads();
    }

    // epilogue: bias + activation
    const int hi = (lane >> 4) & 1;
    const int nl = lane & 15;
#pragma unroll
    for (int tm = 0; tm < 2; tm++)
#pragma unroll
        for (int tn = 0; tn < 2; tn++) {
            int colg = n0 + wn * 32 + tn * 16 + nl;
            if (!AL && colg >= N) continue;
            float bv = bias ? bias[colg] : 0.f;
#pragma unroll
            for (int v = 0; v < 8; v++) {
                int rowg = m0 + wm * 32 + tm * 16 + hi * 8 + v;
                if (!AL && rowg >= M) continue;
                float xv = acc[tm][tn][v] + bv;
                if (act == 1) xv = fmaxf(xv, 0.f);
                else if (act == 2) xv = 0.5f * xv * (1.f + erff(xv * 0.70710678118f));
                D[(long long)rowg * N + colg] = xv;
            }
        }
}

// ---------------- RMA cross-attention: per head (S=64, d=64), full-WMMA in LDS ----------------
__global__ __launch_bounds__(128) void k_rma_attn(
    const float* __restrict__ kk, const float* __restrict__ vv, float* __restrict__ out)
{
    __shared__ bf16_t kkL[64 * 64];   // (i, d)
    __shared__ bf16_t vjd[64 * 64];   // (j, d)  -> B for sim (col=j, K=d contiguous)
    __shared__ bf16_t vdj[64 * 64];   // (d, j)  -> B for out (col=d, K=j contiguous)
    __shared__ float  simL[64 * 64];
    __shared__ bf16_t attnL[64 * 64];

    const int head = blockIdx.x;          // bcl*8 + h (within chunk)
    const int bcl = head >> 3, h = head & 7;
    const float* kkp = kk + ((long long)bcl * 64) * 512 + h * 64;
    const float* vvp = vv + ((long long)bcl * 64) * 512 + h * 64;
    const int tid = threadIdx.x, lane = tid & 31, wave = tid >> 5;

    // vectorized staging: 4096 elems, 8x float4 per thread
    for (int i = tid * 4; i < 4096; i += 128 * 4) {
        int r = i >> 6, d = i & 63;
        const float4 kf = *(const float4*)(kkp + (long long)r * 512 + d);
        const float4 vf = *(const float4*)(vvp + (long long)r * 512 + d);
        v4bf kh, vh;
        kh[0] = (bf16_t)kf.x; kh[1] = (bf16_t)kf.y; kh[2] = (bf16_t)kf.z; kh[3] = (bf16_t)kf.w;
        vh[0] = (bf16_t)vf.x; vh[1] = (bf16_t)vf.y; vh[2] = (bf16_t)vf.z; vh[3] = (bf16_t)vf.w;
        *(v4bf*)(kkL + i) = kh;
        *(v4bf*)(vjd + i) = vh;
        vdj[(d + 0) * 64 + r] = (bf16_t)vf.x;
        vdj[(d + 1) * 64 + r] = (bf16_t)vf.y;
        vdj[(d + 2) * 64 + r] = (bf16_t)vf.z;
        vdj[(d + 3) * 64 + r] = (bf16_t)vf.w;
    }
    __syncthreads();

    // sim = kk * vv^T  (each wave: 16 rows x 64 cols, K=64)
    v8f s[4] = {};
#pragma unroll
    for (int ks = 0; ks < 2; ks++) {
        v16bf af = load_a_frag(kkL, wave * 16, 64, ks * 32, lane);
#pragma unroll
        for (int jt = 0; jt < 4; jt++) {
            v16bf bf_ = load_b_frag(vjd, jt * 16, 64, ks * 32, lane);
            s[jt] = wmma_bf16(af, bf_, s[jt]);
        }
    }
    const int hi = (lane >> 4) & 1, nl = lane & 15;
#pragma unroll
    for (int jt = 0; jt < 4; jt++)
#pragma unroll
        for (int v = 0; v < 8; v++)
            simL[(wave * 16 + hi * 8 + v) * 64 + jt * 16 + nl] = s[jt][v] * 0.125f; // 1/sqrt(64)
    __syncthreads();

    // fp32 softmax over j (one thread per row)
    if (tid < 64) {
        float mx = -1e30f;
        for (int j = 0; j < 64; j++) mx = fmaxf(mx, simL[tid * 64 + j]);
        float sum = 0.f;
        for (int j = 0; j < 64; j++) { float e = __expf(simL[tid * 64 + j] - mx); simL[tid * 64 + j] = e; sum += e; }
        float inv = 1.f / sum;
        for (int j = 0; j < 64; j++) attnL[tid * 64 + j] = (bf16_t)(simL[tid * 64 + j] * inv);
    }
    __syncthreads();

    // out = attn * vv
    v8f o[4] = {};
#pragma unroll
    for (int ks = 0; ks < 2; ks++) {
        v16bf af = load_a_frag(attnL, wave * 16, 64, ks * 32, lane);
#pragma unroll
        for (int dt = 0; dt < 4; dt++) {
            v16bf bf_ = load_b_frag(vdj, dt * 16, 64, ks * 32, lane);
            o[dt] = wmma_bf16(af, bf_, o[dt]);
        }
    }
    float* op = out + ((long long)bcl * 64) * 512 + h * 64;
#pragma unroll
    for (int dt = 0; dt < 4; dt++)
#pragma unroll
        for (int v = 0; v < 8; v++)
            op[(long long)(wave * 16 + hi * 8 + v) * 512 + dt * 16 + nl] = o[dt][v];
}

// ---------------- head-dim-8 MHA (time / feature transformers), online softmax ----------------
// tokens laid out (b,k,l) x C=64; qkv is tokens x 192 (q|k|v).
__global__ __launch_bounds__(64) void k_mha_d8(
    const float* __restrict__ qkv, float* __restrict__ out, int mode, int S)
{
    __shared__ float Ks[192 * 8];
    __shared__ float Vs[192 * 8];
    const int blk = blockIdx.x;
    const int g = blk >> 3, h = blk & 7;
    long long base; int stride;
    if (mode == 0) { base = (long long)g * 192; stride = 1; }            // time: g=(b,k), S=192
    else { int b = g / 192, l = g % 192; base = (long long)b * 12288 + l; stride = 192; } // feat: S=64
    const int tid = threadIdx.x;
    for (int i = tid; i < S * 8; i += 64) {
        int s = i >> 3, d = i & 7;
        long long r = base + (long long)s * stride;
        Ks[i] = qkv[r * 192 + 64 + h * 8 + d];
        Vs[i] = qkv[r * 192 + 128 + h * 8 + d];
    }
    __syncthreads();
    const float sc = 0.35355339059f; // 1/sqrt(8)
    for (int sq = tid; sq < S; sq += 64) {
        long long r = base + (long long)sq * stride;
        float q[8];
#pragma unroll
        for (int d = 0; d < 8; d++) q[d] = qkv[r * 192 + h * 8 + d];
        float m = -1e30f, l = 0.f, acc[8] = {0, 0, 0, 0, 0, 0, 0, 0};
        for (int j = 0; j < S; j++) {
            float s = 0.f;
#pragma unroll
            for (int d = 0; d < 8; d++) s += q[d] * Ks[j * 8 + d];
            s *= sc;
            float mn = fmaxf(m, s);
            float corr = __expf(m - mn);
            float p = __expf(s - mn);
            l = l * corr + p;
#pragma unroll
            for (int d = 0; d < 8; d++) acc[d] = acc[d] * corr + p * Vs[j * 8 + d];
            m = mn;
        }
        float inv = 1.f / l;
#pragma unroll
        for (int d = 0; d < 8; d++) out[r * 64 + h * 8 + d] = acc[d] * inv;
    }
}

// ---------------- small / elementwise kernels ----------------
__global__ __launch_bounds__(128) void k_embed(
    const int* __restrict__ step, const float* __restrict__ d1w, const float* __restrict__ d1b,
    const float* __restrict__ d2w, const float* __restrict__ d2b, float* __restrict__ e)
{
    __shared__ float s0[128], s1[128];
    const int b = blockIdx.x, c = threadIdx.x;
    float st = (float)step[b];
    int j = (c < 64) ? c : c - 64;
    float f = powf(10.f, (float)j * (4.f / 63.f));
    float t = st * f;
    s0[c] = (c < 64) ? sinf(t) : cosf(t);
    __syncthreads();
    float a = 0.f;
    for (int i = 0; i < 128; i++) a += s0[i] * d1w[c * 128 + i];
    a += d1b[c];
    s1[c] = a / (1.f + __expf(-a));
    __syncthreads();
    float a2 = 0.f;
    for (int i = 0; i < 128; i++) a2 += s1[i] * d2w[c * 128 + i];
    a2 += d2b[c];
    e[b * 128 + c] = a2 / (1.f + __expf(-a2));
}

__global__ __launch_bounds__(64) void k_dproj(
    const float* __restrict__ e, const float* __restrict__ dp_w,
    const float* __restrict__ dp_b, float* __restrict__ dproj)
{
    const int blk = blockIdx.x;           // layer*16 + b
    const int i = blk >> 4, b = blk & 15;
    const int c = threadIdx.x;
    const float* w = dp_w + ((long long)i * 64 + c) * 128;
    const float* eb = e + b * 128;
    float a = 0.f;
    for (int t = 0; t < 128; t++) a += eb[t] * w[t];
    dproj[blk * 64 + c] = a + dp_b[i * 64 + c];
}

__global__ void k_in_conv(const float* __restrict__ x, const float* __restrict__ w,
                          const float* __restrict__ bias, float* __restrict__ x4)
{
    long long idx = (long long)blockIdx.x * 256 + threadIdx.x;
    if (idx >= 12582912LL) return;
    long long t = idx >> 6; int c = (int)(idx & 63);
    int b = (int)(t / 12288); long long n = t % 12288;
    float v = w[c * 2 + 0] * x[(long long)b * 24576 + n]
            + w[c * 2 + 1] * x[(long long)b * 24576 + 12288 + n] + bias[c];
    x4[idx] = fmaxf(v, 0.f);
}

__global__ __launch_bounds__(256) void k_cond_t(const float* __restrict__ cond, float* __restrict__ ct)
{
    __shared__ float tile[32][33];
    const int b = blockIdx.z;
    const int n0 = blockIdx.x * 32, s0 = blockIdx.y * 32;
    const float* src = cond + (long long)b * 144 * 12288;
    float* dst = ct + (long long)b * 12288 * 144;
    const int tx = threadIdx.x & 31, ty = threadIdx.x >> 5;
    for (int i = ty; i < 32; i += 8) {
        int s = s0 + i, n = n0 + tx;
        tile[i][tx] = (s < 144) ? src[(long long)s * 12288 + n] : 0.f;
    }
    __syncthreads();
    for (int i = ty; i < 32; i += 8) {
        int n = n0 + i, s = s0 + tx;
        if (s < 144) dst[(long long)n * 144 + s] = tile[tx][i];
    }
}

__global__ void k_add_dproj(const float* __restrict__ x4, const float* __restrict__ dproj,
                            float* __restrict__ y, int layer)
{
    long long idx = (long long)blockIdx.x * 256 + threadIdx.x;
    if (idx >= 12582912LL) return;
    long long t = idx >> 6; int c = (int)(idx & 63);
    int b = (int)(t / 12288);
    y[idx] = x4[idx] + dproj[((long long)layer * 16 + b) * 64 + c];
}

// build packed A-matrices for kk/vv GEMMs: row = (bc_local)*64 + k, cols = [y | ci | ref] / [y | ref]
__global__ __launch_bounds__(128) void k_pack(
    const float* __restrict__ y, const float* __restrict__ ci, const float* __restrict__ ref,
    float* __restrict__ pk, float* __restrict__ pv, int bc0)
{
    const int r = blockIdx.x;                 // 0..16383
    const int bc = bc0 + (r >> 6), k = r & 63;
    const int b = bc >> 6, c = bc & 63;
    const long long tb = (((long long)(b * 64 + k)) * 192) * 64 + c;  // y[tb + l*64]
    const float* refp = ref + ((long long)(b * 64 + k)) * 288;
    float* pkr = pk + (long long)r * 672;
    float* pvr = pv + (long long)r * 480;
    for (int col = threadIdx.x; col < 672; col += 128) {
        float v;
        if (col < 192)       v = y[tb + (long long)col * 64];
        else if (col < 384)  v = ci[tb + (long long)(col - 192) * 64];
        else                 v = refp[col - 384];
        pkr[col] = v;
    }
    for (int col = threadIdx.x; col < 480; col += 128) {
        float v;
        if (col < 192) v = y[tb + (long long)col * 64];
        else           v = refp[col - 192];
        pvr[col] = v;
    }
}

__global__ void k_add_rma(const float* __restrict__ rmo, float* __restrict__ y, int bc0)
{
    long long idx = (long long)blockIdx.x * 256 + threadIdx.x;   // 16384*192
    if (idx >= 3145728LL) return;
    int r = (int)(idx / 192), l = (int)(idx % 192);
    int bc = bc0 + (r >> 6), k = r & 63;
    int b = bc >> 6, c = bc & 63;
    y[(((long long)(b * 64 + k)) * 192 + l) * 64 + c] += rmo[idx];
}

__global__ __launch_bounds__(64) void k_add_ln(
    const float* __restrict__ x, const float* __restrict__ r,
    const float* __restrict__ g, const float* __restrict__ bta, float* __restrict__ out)
{
    const long long row = blockIdx.x;
    const int c = threadIdx.x;
    float v = x[row * 64 + c] + r[row * 64 + c];
    __shared__ float red[64];
    red[c] = v; __syncthreads();
    for (int off = 32; off > 0; off >>= 1) { if (c < off) red[c] += red[c + off]; __syncthreads(); }
    float mean = red[0] * (1.f / 64.f);
    __syncthreads();
    float d = v - mean;
    red[c] = d * d; __syncthreads();
    for (int off = 32; off > 0; off >>= 1) { if (c < off) red[c] += red[c + off]; __syncthreads(); }
    float var = red[0] * (1.f / 64.f);
    out[row * 64 + c] = d * rsqrtf(var + 1e-5f) * g[c] + bta[c];
}

__global__ void k_glu(const float* __restrict__ b2, float* __restrict__ out)
{
    long long idx = (long long)blockIdx.x * 256 + threadIdx.x;
    if (idx >= 12582912LL) return;
    long long t = idx >> 6; int c = (int)(idx & 63);
    float g = b2[t * 128 + c], f = b2[t * 128 + 64 + c];
    out[idx] = (1.f / (1.f + __expf(-g))) * tanhf(f);
}

__global__ void k_res_skip(const float* __restrict__ b2, float* __restrict__ x4, float* __restrict__ skip)
{
    long long idx = (long long)blockIdx.x * 256 + threadIdx.x;
    if (idx >= 12582912LL) return;
    long long t = idx >> 6; int c = (int)(idx & 63);
    x4[idx] = (x4[idx] + b2[t * 128 + c]) * 0.70710678118f;
    skip[idx] += b2[t * 128 + 64 + c];
}

// ---------------- host orchestration ----------------
extern "C" void kernel_launch(void* const* d_in, const int* in_sizes, int n_in,
                              void* d_out, int out_size, void* d_ws, size_t ws_size,
                              hipStream_t stream)
{
    const float* x     = (const float*)d_in[0];
    const float* cond  = (const float*)d_in[1];
    const int*   dstep = (const int*)  d_in[2];
    const float* ref   = (const float*)d_in[3];
    const float* in_w  = (const float*)d_in[4];
    const float* in_b  = (const float*)d_in[5];
    const float* d1w = (const float*)d_in[6],  *d1b = (const float*)d_in[7];
    const float* d2w = (const float*)d_in[8],  *d2b = (const float*)d_in[9];
    const float* o1w = (const float*)d_in[10], *o1b = (const float*)d_in[11];
    const float* o2w = (const float*)d_in[12], *o2b = (const float*)d_in[13];
    const float* dp_w = (const float*)d_in[14], *dp_b = (const float*)d_in[15];
    const float* cp_w = (const float*)d_in[16], *cp_b = (const float*)d_in[17];
    const float* mid_w = (const float*)d_in[18], *mid_b = (const float*)d_in[19];
    const float* op_w  = (const float*)d_in[20], *op_b  = (const float*)d_in[21];
    const float* rk_w = (const float*)d_in[22], *rv_w = (const float*)d_in[23];
    const float* ro_w = (const float*)d_in[24], *ro_b = (const float*)d_in[25];
    const float* tw[12]; const float* fw[12];
    for (int i = 0; i < 12; i++) { tw[i] = (const float*)d_in[26 + i]; fw[i] = (const float*)d_in[38 + i]; }
    float* out = (float*)d_out;
    float* ws  = (float*)d_ws;

    const long long M  = 196608;  // B*K*L tokens
    const long long RR = 16384;   // RMA chunk rows (256 bc * 64 k)

    size_t off = 0;
    auto take = [&](size_t n) { size_t r = off; off += (n + 63) & ~(size_t)63; return r; };
    float* pE    = ws + take(16 * 128);
    float* pDP   = ws + take(4 * 16 * 64);
    float* pCT   = ws + take((size_t)M * 144);
    float* pX4   = ws + take((size_t)M * 64);
    float* pY    = ws + take((size_t)M * 64);
    float* pCI   = ws + take((size_t)M * 64);
    float* pSKIP = ws + take((size_t)M * 64);
    // scratch union (transformer phase layout; RMA chunk buffers overlay the same region)
    float* pQKV = ws + off;                       // M*192
    float* pATT = pQKV + (size_t)M * 192;         // M*64
    float* pT1  = pATT + (size_t)M * 64;          // M*64
    float* pT2  = pT1  + (size_t)M * 64;          // M*64
    float* pB2  = pQKV;                           // M*128 (after qkv is dead)
    float* pPK  = pQKV;                           // RR*672
    float* pPV  = pPK + RR * 672;                 // RR*480
    float* pKK  = pPV + RR * 480;                 // RR*512
    float* pVV  = pKK + RR * 512;                 // RR*512
    float* pAO  = pVV + RR * 512;                 // RR*512
    float* pRO  = pAO + RR * 512;                 // RR*192

    auto gemm = [&](const float* A, const float* W, const float* bias, float* D,
                    long long Mm, int N, int K, int act, float ascale) {
        dim3 g((unsigned)((N + BN - 1) / BN), (unsigned)((Mm + BM - 1) / BM), 1);
        bool aligned = (Mm % BM == 0) && (N % BN == 0) && (K % BK == 0);
        if (aligned)
            gemm_bf16<true><<<g, 256, 0, stream>>>(A, W, bias, D, (int)Mm, N, K, act, ascale);
        else
            gemm_bf16<false><<<g, 256, 0, stream>>>(A, W, bias, D, (int)Mm, N, K, act, ascale);
    };

    auto encoder = [&](const float* const* w, int mode) {
        gemm(pY, w[0], w[1], pQKV, M, 192, 64, 0, 1.f);                     // qkv
        int nblk = (mode == 0 ? 1024 : 3072) * 8;
        int S = (mode == 0) ? 192 : 64;
        k_mha_d8<<<nblk, 64, 0, stream>>>(pQKV, pATT, mode, S);             // attention
        gemm(pATT, w[2], w[3], pT1, M, 64, 64, 0, 1.f);                     // o-proj
        k_add_ln<<<(unsigned)M, 64, 0, stream>>>(pY, pT1, w[8], w[9], pY);  // ln1
        gemm(pY, w[4], w[5], pT1, M, 64, 64, 2, 1.f);                       // ff1 + gelu
        gemm(pT1, w[6], w[7], pT2, M, 64, 64, 0, 1.f);                      // ff2
        k_add_ln<<<(unsigned)M, 64, 0, stream>>>(pY, pT2, w[10], w[11], pY);// ln2
    };

    const unsigned EW = (unsigned)((M * 64 + 255) / 256);

    hipMemsetAsync(pSKIP, 0, (size_t)M * 64 * sizeof(float), stream);
    k_embed<<<16, 128, 0, stream>>>(dstep, d1w, d1b, d2w, d2b, pE);
    k_dproj<<<64, 64, 0, stream>>>(pE, dp_w, dp_b, pDP);
    k_in_conv<<<EW, 256, 0, stream>>>(x, in_w, in_b, pX4);
    { dim3 g(384, 5, 16); k_cond_t<<<g, 256, 0, stream>>>(cond, pCT); }

    for (int i = 0; i < 4; i++) {
        k_add_dproj<<<EW, 256, 0, stream>>>(pX4, pDP, pY, i);
        gemm(pCT, cp_w + (size_t)i * 64 * 144, cp_b + i * 64, pCI, M, 64, 144, 0, 1.f);

        // Reference-modulated cross attention, chunked over bc
        for (int ch = 0; ch < 4; ch++) {
            int bc0 = ch * 256;
            k_pack<<<16384, 128, 0, stream>>>(pY, pCI, ref, pPK, pPV, bc0);
            gemm(pPK, rk_w + (size_t)i * 512 * 672, nullptr, pKK, RR, 512, 672, 0, 1.f);
            gemm(pPV, rv_w + (size_t)i * 512 * 480, nullptr, pVV, RR, 512, 480, 0, 1.f);
            k_rma_attn<<<2048, 128, 0, stream>>>(pKK, pVV, pAO);
            gemm(pAO, ro_w + (size_t)i * 192 * 512, ro_b + (size_t)i * 192, pRO, RR, 192, 512, 0, 1.f);
            k_add_rma<<<12288, 256, 0, stream>>>(pRO, pY, bc0);
        }

        const float* tp[12] = {
            tw[0] + (size_t)i * 192 * 64, tw[1] + i * 192, tw[2] + (size_t)i * 64 * 64, tw[3] + i * 64,
            tw[4] + (size_t)i * 64 * 64, tw[5] + i * 64, tw[6] + (size_t)i * 64 * 64, tw[7] + i * 64,
            tw[8] + i * 64, tw[9] + i * 64, tw[10] + i * 64, tw[11] + i * 64 };
        encoder(tp, 0);   // time transformer over L
        const float* fp[12] = {
            fw[0] + (size_t)i * 192 * 64, fw[1] + i * 192, fw[2] + (size_t)i * 64 * 64, fw[3] + i * 64,
            fw[4] + (size_t)i * 64 * 64, fw[5] + i * 64, fw[6] + (size_t)i * 64 * 64, fw[7] + i * 64,
            fw[8] + i * 64, fw[9] + i * 64, fw[10] + i * 64, fw[11] + i * 64 };
        encoder(fp, 1);   // feature transformer over K

        gemm(pY, mid_w + (size_t)i * 128 * 64, mid_b + i * 128, pB2, M, 128, 64, 0, 1.f);
        k_glu<<<EW, 256, 0, stream>>>(pB2, pT1);
        gemm(pT1, op_w + (size_t)i * 128 * 64, op_b + i * 128, pB2, M, 128, 64, 0, 1.f);
        k_res_skip<<<EW, 256, 0, stream>>>(pB2, pX4, pSKIP);
    }

    // output head: 1/sqrt(NL) folded into A-scale of o1
    gemm(pSKIP, o1w, o1b, pT1, M, 64, 64, 1, 0.5f);
    gemm(pT1, o2w, o2b, out, M, 1, 64, 0, 1.f);
    (void)in_sizes; (void)n_in; (void)out_size; (void)ws_size;
}
